// TaskGNN_62500364091967
// MI455X (gfx1250) — compile-verified, matrix operand order checked by reading, not run on hardware
//
#include <hip/hip_runtime.h>

typedef _Float16 v16h __attribute__((ext_vector_type(16)));
typedef _Float16 v8h  __attribute__((ext_vector_type(8)));
typedef float    v8f  __attribute__((ext_vector_type(8)));

#define NEG_SLOPE 0.2f
#define NEG_INF_KEY ((int)0x807FFFFFu)   // monotone key of -inf

// ---- monotone float<->int key so atomicMax(int) == float max ----
__device__ __forceinline__ int f32_key(float f) {
    int i = __float_as_int(f);
    return i < 0 ? (i ^ 0x7FFFFFFF) : i;
}
__device__ __forceinline__ float key_f32(int k) {
    int i = k < 0 ? (k ^ 0x7FFFFFFF) : k;
    return __int_as_float(i);
}

// ---------------- fills ----------------
__global__ void k_fill_f32(float* p, float v, long long n) {
    long long i = (long long)blockIdx.x * blockDim.x + threadIdx.x;
    if (i < n) p[i] = v;
}
__global__ void k_fill_i32(int* p, int v, long long n) {
    long long i = (long long)blockIdx.x * blockDim.x + threadIdx.x;
    if (i < n) p[i] = v;
}

// ---------------- pack W2 [1024,128] f32 -> B-fragment order f16 ----------------
// frag[((t*32+s)*32+lane)*16 + e], t=colTile(8), s=kStep(32), lane(32)
__global__ void k_packw2(const float* __restrict__ W2, _Float16* __restrict__ w2f) {
    int id = blockIdx.x * blockDim.x + threadIdx.x;
    if (id >= 8 * 32 * 32) return;
    int l = id & 31, s = (id >> 5) & 31, t = id >> 10;
    int col = t * 16 + (l & 15);
    int kb  = s * 32 + ((l >> 4) ? 8 : 0);
    _Float16* o = w2f + (size_t)id * 16;
    for (int e = 0; e < 16; ++e) {
        int K = kb + (e < 8 ? e : 8 + e);   // {0..7} then {16..23} relative
        o[e] = (_Float16)W2[(size_t)K * 128 + col];
    }
}

// ---------------- layer-1 GEMM: h1 = x[N,4] @ W1[4,1024] via WMMA (K padded 4->32) ----------------
__global__ void k_gemm1(const float* __restrict__ x, const float* __restrict__ W1,
                        _Float16* __restrict__ h16, int N) {
    int wave = blockIdx.x * (blockDim.x >> 5) + (threadIdx.x >> 5);
    int lane = threadIdx.x & 31;
    int ntm = (N + 15) >> 4;
    if (wave >= ntm * 64) return;            // wave-uniform exit
    int tm = wave >> 6, tn = wave & 63;
    int row = tm * 16 + (lane & 15);
    if (row >= N) row = N - 1;
    int col = tn * 16 + (lane & 15);

    v16h a, b;
    for (int i = 0; i < 16; ++i) { a[i] = (_Float16)0.f; b[i] = (_Float16)0.f; }
    if (lane < 16) {                          // only K=0..3 are live
        const float* xr = x + (size_t)row * 4;
        a[0] = (_Float16)xr[0]; a[1] = (_Float16)xr[1];
        a[2] = (_Float16)xr[2]; a[3] = (_Float16)xr[3];
        b[0] = (_Float16)W1[0 * 1024 + col]; b[1] = (_Float16)W1[1 * 1024 + col];
        b[2] = (_Float16)W1[2 * 1024 + col]; b[3] = (_Float16)W1[3 * 1024 + col];
    }
    v8f c = {};
    c = __builtin_amdgcn_wmma_f32_16x16x32_f16(false, a, false, b, (short)0, c, false, false);

    int nc = lane & 15, mb = (lane >> 4) * 8;
    for (int r = 0; r < 8; ++r) {
        int m = tm * 16 + mb + r;
        if (m < N) h16[(size_t)m * 1024 + tn * 16 + nc] = (_Float16)c[r];
    }
}

// ---------------- layer-2 GEMM: h2 = h1f16[N,1024] @ W2f16[1024,128] ----------------
__global__ void k_gemm2(const _Float16* __restrict__ hin, const _Float16* __restrict__ w2f,
                        float* __restrict__ h2, int N) {
    int wave = blockIdx.x * (blockDim.x >> 5) + (threadIdx.x >> 5);
    int lane = threadIdx.x & 31;
    int ntm = (N + 15) >> 4;
    if (wave >= ntm * 8) return;
    int tm = wave >> 3, tn = wave & 7;
    int arow = tm * 16 + (lane & 15);
    if (arow >= N) arow = N - 1;
    const _Float16* ap = hin + (size_t)arow * 1024 + ((lane >> 4) ? 8 : 0);
    const _Float16* bp = w2f + ((size_t)(tn * 32) * 32 + lane) * 16;

    v8f c = {};
    for (int s = 0; s < 32; ++s) {
        const v8h* pa = (const v8h*)(ap + s * 32);
        v8h a0 = pa[0];                      // K = kb + {0..7}
        v8h a1 = pa[2];                      // K = kb + {16..23}
        v16h a;
        for (int i = 0; i < 8; ++i) { a[i] = a0[i]; a[8 + i] = a1[i]; }
        v16h b = *(const v16h*)(bp + (size_t)s * 32 * 16);
        c = __builtin_amdgcn_wmma_f32_16x16x32_f16(false, a, false, b, (short)0, c, false, false);
    }
    int nc = lane & 15, mb = (lane >> 4) * 8;
    for (int r = 0; r < 8; ++r) {
        int m = tm * 16 + mb + r;
        if (m < N) h2[(size_t)m * 128 + tn * 16 + nc] = c[r];
    }
}

// ---------------- per-node attention coefficients ----------------
__global__ void k_alpha1(const _Float16* __restrict__ h16, const float* __restrict__ as,
                         const float* __restrict__ ad, float* oa, float* ob, int N) {
    int id = blockIdx.x * blockDim.x + threadIdx.x;
    if (id >= N * 8) return;
    int n = id >> 3, h = id & 7;
    const _Float16* p = h16 + (size_t)n * 1024 + h * 128;
    const float *pa = as + h * 128, *pb = ad + h * 128;
    float sa = 0.f, sb = 0.f;
    for (int c = 0; c < 128; ++c) { float v = (float)p[c]; sa += v * pa[c]; sb += v * pb[c]; }
    oa[id] = sa; ob[id] = sb;
}
__global__ void k_alpha2(const float* __restrict__ h2, const float* __restrict__ as,
                         const float* __restrict__ ad, float* oa, float* ob, int N) {
    int n = blockIdx.x * blockDim.x + threadIdx.x;
    if (n >= N) return;
    const float* p = h2 + (size_t)n * 128;
    float sa = 0.f, sb = 0.f;
    for (int c = 0; c < 128; ++c) { sa += p[c] * as[c]; sb += p[c] * ad[c]; }
    oa[n] = sa; ob[n] = sb;
}

// ---------------- segment softmax pass A: atomic max of leaky-relu scores ----------------
__global__ void k_edge_max(const int* __restrict__ ei, const float* __restrict__ as,
                           const float* __restrict__ ad, int* __restrict__ emax,
                           int E, int N, int H) {
    long long id = (long long)blockIdx.x * blockDim.x + threadIdx.x;
    if (id >= (long long)(E + N) * H) return;
    int h = (int)(id % H); long long e = id / H;
    int s, d;
    if (e < E) { s = ei[e]; d = ei[(size_t)E + e]; } else { s = d = (int)(e - E); }
    float v = as[(size_t)s * H + h] + ad[(size_t)d * H + h];
    v = v > 0.f ? v : NEG_SLOPE * v;
    atomicMax(emax + (size_t)d * H + h, f32_key(v));
}

// ---------------- pass B: p = exp(e - max), atomic denom ----------------
__global__ void k_edge_exp(const int* __restrict__ ei, const float* __restrict__ as,
                           const float* __restrict__ ad, const int* __restrict__ emax,
                           float* __restrict__ p, float* __restrict__ denom,
                           int E, int N, int H) {
    long long id = (long long)blockIdx.x * blockDim.x + threadIdx.x;
    if (id >= (long long)(E + N) * H) return;
    int h = (int)(id % H); long long e = id / H;
    int s, d;
    if (e < E) { s = ei[e]; d = ei[(size_t)E + e]; } else { s = d = (int)(e - E); }
    float v = as[(size_t)s * H + h] + ad[(size_t)d * H + h];
    v = v > 0.f ? v : NEG_SLOPE * v;
    float pe = __expf(v - key_f32(emax[(size_t)d * H + h]));
    p[id] = pe;
    atomicAdd(denom + (size_t)d * H + h, pe);
}

// ---------------- pass C layer 1: acc[dst,h,:] += alpha * h16[src,h,:]  (wave per edge-head) ----------------
__global__ void k_agg1(const int* __restrict__ ei, const _Float16* __restrict__ h16,
                       const float* __restrict__ p, const float* __restrict__ denom,
                       float* __restrict__ acc, int E, int N) {
    int lane = threadIdx.x & 31;
    long long wave = (long long)blockIdx.x * (blockDim.x >> 5) + (threadIdx.x >> 5);
    if (wave >= (long long)(E + N) * 8) return;
    int h = (int)(wave & 7); long long e = wave >> 3;
    int s, d;
    if (e < E) { s = ei[e]; d = ei[(size_t)E + e]; } else { s = d = (int)(e - E); }
    float alpha = p[(size_t)e * 8 + h] / (denom[(size_t)d * 8 + h] + 1e-16f);
    const _Float16* hp = h16 + (size_t)s * 1024 + h * 128 + lane * 4;
    float* op = acc + (size_t)d * 1024 + h * 128 + lane * 4;
    for (int j = 0; j < 4; ++j) atomicAdd(op + j, alpha * (float)hp[j]);
}

// ---------------- pass C layer 2 (H=1, into d_out) ----------------
__global__ void k_agg2(const int* __restrict__ ei, const float* __restrict__ h2,
                       const float* __restrict__ p, const float* __restrict__ denom,
                       float* __restrict__ out, int E, int N) {
    int lane = threadIdx.x & 31;
    long long wave = (long long)blockIdx.x * (blockDim.x >> 5) + (threadIdx.x >> 5);
    if (wave >= (long long)(E + N)) return;
    long long e = wave;
    int s, d;
    if (e < E) { s = ei[e]; d = ei[(size_t)E + e]; } else { s = d = (int)(e - E); }
    float alpha = p[e] / (denom[d] + 1e-16f);
    const float* hp = h2 + (size_t)s * 128 + lane * 4;
    float* op = out + (size_t)d * 128 + lane * 4;
    for (int j = 0; j < 4; ++j) atomicAdd(op + j, alpha * hp[j]);
}

// ---------------- post layer 1: relu(acc+b1) -> f16 (feeds layer-2 WMMA) ----------------
__global__ void k_postl1(const float* __restrict__ acc, const float* __restrict__ b1,
                         _Float16* __restrict__ h16, long long n) {
    long long i = (long long)blockIdx.x * blockDim.x + threadIdx.x;
    if (i >= n) return;
    float v = acc[i] + b1[i & 1023];
    h16[i] = (_Float16)(v > 0.f ? v : 0.f);
}
// ---------------- post layer 2: out = relu(out + b2) in place ----------------
__global__ void k_postl2(float* __restrict__ out, const float* __restrict__ b2, long long n) {
    long long i = (long long)blockIdx.x * blockDim.x + threadIdx.x;
    if (i >= n) return;
    float v = out[i] + b2[i & 127];
    out[i] = v > 0.f ? v : 0.f;
}

static inline unsigned nblk(long long n, int t) { return (unsigned)((n + t - 1) / t); }

extern "C" void kernel_launch(void* const* d_in, const int* in_sizes, int n_in,
                              void* d_out, int out_size, void* d_ws, size_t ws_size,
                              hipStream_t stream) {
    const float* x   = (const float*)d_in[0];
    const int*   ei  = (const int*)d_in[1];
    const float* W1  = (const float*)d_in[2];
    const float* as1 = (const float*)d_in[3];
    const float* ad1 = (const float*)d_in[4];
    const float* b1  = (const float*)d_in[5];
    const float* W2  = (const float*)d_in[6];
    const float* as2 = (const float*)d_in[7];
    const float* ad2 = (const float*)d_in[8];
    const float* b2  = (const float*)d_in[9];
    float* out = (float*)d_out;

    const int N  = in_sizes[0] / 4;      // D_IN = 4
    const int E  = in_sizes[1] / 2;      // edge_index [2,E]
    const int ET = E + N;                // with self loops

    char* ws = (char*)d_ws;
    size_t off = 0;
    auto alloc = [&](size_t bytes) -> char* {
        char* r = ws + off;
        off = (off + bytes + 255) & ~(size_t)255;
        return r;
    };
    _Float16* h16 = (_Float16*)alloc((size_t)N * 1024 * 2);
    float* acc1   = (float*)alloc((size_t)N * 1024 * 4);
    float* h2     = (float*)alloc((size_t)N * 128 * 4);
    float* A1s    = (float*)alloc((size_t)N * 8 * 4);
    float* A1d    = (float*)alloc((size_t)N * 8 * 4);
    int*   M1     = (int*)alloc((size_t)N * 8 * 4);
    float* D1     = (float*)alloc((size_t)N * 8 * 4);
    float* P1     = (float*)alloc((size_t)ET * 8 * 4);
    float* A2s    = (float*)alloc((size_t)N * 4);
    float* A2d    = (float*)alloc((size_t)N * 4);
    int*   M2     = (int*)alloc((size_t)N * 4);
    float* D2     = (float*)alloc((size_t)N * 4);
    float* P2     = (float*)alloc((size_t)ET * 4);
    _Float16* W2F = (_Float16*)alloc((size_t)8 * 32 * 32 * 16 * 2);
    (void)ws_size; (void)n_in; (void)out_size;

    const int T = 256;
    // init accumulators / softmax state
    k_fill_i32<<<nblk((long long)N * 8, T), T, 0, stream>>>(M1, NEG_INF_KEY, (long long)N * 8);
    k_fill_f32<<<nblk((long long)N * 8, T), T, 0, stream>>>(D1, 0.f, (long long)N * 8);
    k_fill_f32<<<nblk((long long)N * 1024, T), T, 0, stream>>>(acc1, 0.f, (long long)N * 1024);
    k_fill_i32<<<nblk(N, T), T, 0, stream>>>(M2, NEG_INF_KEY, N);
    k_fill_f32<<<nblk(N, T), T, 0, stream>>>(D2, 0.f, N);
    k_fill_f32<<<nblk((long long)N * 128, T), T, 0, stream>>>(out, 0.f, (long long)N * 128);
    k_packw2<<<nblk(8 * 32 * 32, T), T, 0, stream>>>(W2, W2F);

    // ---- layer 1 ----
    long long tiles1 = (long long)((N + 15) / 16) * 64;
    k_gemm1<<<nblk(tiles1 * 32, T), T, 0, stream>>>(x, W1, h16, N);
    k_alpha1<<<nblk((long long)N * 8, T), T, 0, stream>>>(h16, as1, ad1, A1s, A1d, N);
    k_edge_max<<<nblk((long long)ET * 8, T), T, 0, stream>>>(ei, A1s, A1d, M1, E, N, 8);
    k_edge_exp<<<nblk((long long)ET * 8, T), T, 0, stream>>>(ei, A1s, A1d, M1, P1, D1, E, N, 8);
    k_agg1<<<nblk((long long)ET * 8 * 32, T), T, 0, stream>>>(ei, h16, P1, D1, acc1, E, N);
    k_postl1<<<nblk((long long)N * 1024, T), T, 0, stream>>>(acc1, b1, h16, (long long)N * 1024);

    // ---- layer 2 ----
    long long tiles2 = (long long)((N + 15) / 16) * 8;
    k_gemm2<<<nblk(tiles2 * 32, T), T, 0, stream>>>(h16, W2F, h2, N);
    k_alpha2<<<nblk(N, T), T, 0, stream>>>(h2, as2, ad2, A2s, A2d, N);
    k_edge_max<<<nblk(ET, T), T, 0, stream>>>(ei, A2s, A2d, M2, E, N, 1);
    k_edge_exp<<<nblk(ET, T), T, 0, stream>>>(ei, A2s, A2d, M2, P2, D2, E, N, 1);
    k_agg2<<<nblk((long long)ET * 32, T), T, 0, stream>>>(ei, h2, P2, D2, out, E, N);
    k_postl2<<<nblk((long long)N * 128, T), T, 0, stream>>>(out, b2, (long long)N * 128);
}
